// mLSTMWeavingLayer_29377576305202
// MI455X (gfx1250) — matrix operand order repaired and checked by reading, not compiled, not run on hardware
//
#include <hip/hip_runtime.h>
#include <hip/hip_bf16.h>
#include <math.h>

// Problem constants (match reference)
// B=2 S=512 D=512 INNER=1024 NH=8 DH=128 K=4 BS=4 NPH=256

typedef __attribute__((ext_vector_type(16))) __bf16 v16bf;
typedef __attribute__((ext_vector_type(8)))  float  v8f;

#define GAS __attribute__((address_space(1)))
#define LAS __attribute__((address_space(3)))

#if defined(__has_builtin)
#if __has_builtin(__builtin_amdgcn_global_load_async_to_lds_b32) && \
    __has_builtin(__builtin_amdgcn_s_wait_asynccnt)
#define USE_ASYNC_LDS 1
#endif
#endif
#ifndef USE_ASYNC_LDS
#define USE_ASYNC_LDS 0
#endif

// ---------------------------------------------------------------------------
// fp32-in / fp32-out GEMM via bf16 WMMA. One wave per 16x64 output strip:
// A fragment loaded once per k-step and reused across 4 WMMAs (4 N-tiles).
// M % 16 == 0, N % 64 == 0, K % 32 == 0 (true: 1024x2048x512, 1024x512x1024).
// ---------------------------------------------------------------------------
__global__ __launch_bounds__(256) void wmma_gemm_kernel(
    const float* __restrict__ A, const float* __restrict__ Bm,
    float* __restrict__ C, int M, int N, int K) {
  const int lane = threadIdx.x & 31;
  const int wave = blockIdx.x * (blockDim.x >> 5) + (threadIdx.x >> 5);
  const int nSuper = N >> 6;
  const int total = (M >> 4) * nSuper;
  if (wave >= total) return;
  const int mT = wave / nSuper, nS = wave % nSuper;
  const int laneLo = lane & 15, laneHi = lane >> 4;
  const float* arowp = A + (size_t)(mT * 16 + laneLo) * K;
  const float* bbase = Bm + nS * 64;
  v8f acc0 = {}, acc1 = {}, acc2 = {}, acc3 = {};
  for (int k0 = 0; k0 < K; k0 += 32) {
    __builtin_prefetch(arowp + k0 + 64, 0, 0);  // global_prefetch_b8
    // A 16x32 bf16 fragment: lane holds row M=lane%16; element e -> K per table
    v16bf a;
#pragma unroll
    for (int v = 0; v < 8; ++v) {
      const int kb = (v < 4 ? 2 * v : 16 + 2 * (v - 4)) + 8 * laneHi;
      const float2 t = *(const float2*)(arowp + k0 + kb);
      a[2 * v]     = (__bf16)t.x;
      a[2 * v + 1] = (__bf16)t.y;
    }
    // B 32x16 fragments x4: lane holds col N=lane%16; element e -> K per table
    const float* bp = bbase + (size_t)k0 * N + laneLo;
    v16bf bt[4];
#pragma unroll
    for (int nt = 0; nt < 4; ++nt) {
#pragma unroll
      for (int e = 0; e < 16; ++e) {
        const int kk = (e < 8 ? e : 8 + e) + 8 * laneHi;
        bt[nt][e] = (__bf16)bp[(size_t)kk * N + nt * 16];
      }
    }
    acc0 = __builtin_amdgcn_wmma_f32_16x16x32_bf16(false, a, false, bt[0],
                                                   (short)0, acc0, false, false);
    acc1 = __builtin_amdgcn_wmma_f32_16x16x32_bf16(false, a, false, bt[1],
                                                   (short)0, acc1, false, false);
    acc2 = __builtin_amdgcn_wmma_f32_16x16x32_bf16(false, a, false, bt[2],
                                                   (short)0, acc2, false, false);
    acc3 = __builtin_amdgcn_wmma_f32_16x16x32_bf16(false, a, false, bt[3],
                                                   (short)0, acc3, false, false);
  }
  // D layout: element j -> row j + 8*laneHi, col laneLo (per 16-wide tile)
#pragma unroll
  for (int j = 0; j < 8; ++j) {
    float* cp = C + (size_t)(mT * 16 + j + 8 * laneHi) * N + nS * 64 + laneLo;
    cp[0]  = acc0[j];
    cp[16] = acc1[j];
    cp[32] = acc2[j];
    cp[48] = acc3[j];
  }
}

// ---------------------------------------------------------------------------
// Depthwise causal conv(K=4) + SiLU, block-diagonal 4x4 q/k/v projections.
// One block per (b,s); thread n handles channel block [4n, 4n+3].
// hh = [x_lstm | z] (B*S, 2048). Writes qkv packed (B*S, 3072) and xa.
// ---------------------------------------------------------------------------
__global__ __launch_bounds__(256) void conv_qkv_kernel(
    const float* __restrict__ hh, const float* __restrict__ conv_w,
    const float* __restrict__ conv_b,
    const float* __restrict__ w_q, const float* __restrict__ w_k,
    const float* __restrict__ w_v,
    float* __restrict__ qkv, float* __restrict__ xa_out) {
  const int bs = blockIdx.x;      // b*512 + s
  const int s  = bs & 511;
  const int n  = threadIdx.x;     // 0..255 = NPH block index
  const int c0 = n * 4;
  float xl[4][4];                 // [tap][chan]; tap 3 == current timestep
#pragma unroll
  for (int t = 0; t < 4; ++t) {
    const int sp = s - 3 + t;
    if (sp >= 0) {
      const float* p = hh + (size_t)(bs - 3 + t) * 2048 + c0;
#pragma unroll
      for (int u = 0; u < 4; ++u) xl[t][u] = p[u];
    } else {
#pragma unroll
      for (int u = 0; u < 4; ++u) xl[t][u] = 0.f;
    }
  }
  float xa4[4];
#pragma unroll
  for (int u = 0; u < 4; ++u) {
    const int c = c0 + u;
    float acc = conv_b[c];
#pragma unroll
    for (int t = 0; t < 4; ++t) acc = fmaf(conv_w[t * 1024 + c], xl[t][u], acc);
    xa4[u] = acc / (1.f + expf(-acc));   // SiLU
  }
  float* out = qkv + (size_t)bs * 3072;
#pragma unroll
  for (int r = 0; r < 4; ++r) {
    float aq = 0.f, ak = 0.f, av = 0.f;
#pragma unroll
    for (int d = 0; d < 4; ++d) {
      const int wi = (n * 4 + r) * 4 + d;   // w[(n,r,d)]
      aq = fmaf(w_q[wi], xa4[d], aq);
      ak = fmaf(w_k[wi], xa4[d], ak);
      av = fmaf(w_v[wi], xl[3][d], av);
    }
    out[c0 + r]        = aq;
    out[1024 + c0 + r] = ak;
    out[2048 + c0 + r] = av;
  }
  float* xo = xa_out + (size_t)bs * 1024 + c0;
#pragma unroll
  for (int u = 0; u < 4; ++u) xo[u] = xa4[u];
}

// ---------------------------------------------------------------------------
// Gate pre-activations: i_pre = qkv@w_i + b_i ; log_f = logsigmoid(qkv@w_f+b_f)
// Block per (b,s); 16 waves, one output scalar per wave (8 i-heads, 8 f-heads).
// ---------------------------------------------------------------------------
__global__ __launch_bounds__(512) void gates_kernel(
    const float* __restrict__ qkv, const float* __restrict__ w_i,
    const float* __restrict__ b_i, const float* __restrict__ w_f,
    const float* __restrict__ b_f, float* __restrict__ ipre,
    float* __restrict__ logfg) {
  const int bs   = blockIdx.x;
  const int lane = threadIdx.x & 31;
  const int w    = threadIdx.x >> 5;   // 0..15
  const int head = w & 7;
  const bool isF = (w >= 8);
  const float* W = isF ? w_f : w_i;    // (3072, 8) row-major
  const float* q = qkv + (size_t)bs * 3072;
  float acc = 0.f;
  for (int j = lane; j < 3072; j += 32) acc = fmaf(q[j], W[j * 8 + head], acc);
#pragma unroll
  for (int off = 16; off > 0; off >>= 1) acc += __shfl_xor(acc, off, 32);
  if (lane == 0) {
    if (!isF) {
      ipre[bs * 8 + head] = acc + b_i[head];
    } else {
      const float p = acc + b_f[head];
      // stable log_sigmoid
      logfg[bs * 8 + head] = fminf(p, 0.f) - log1pf(expf(-fabsf(p)));
    }
  }
}

// ---------------------------------------------------------------------------
// Sequential mLSTM scan. One block per (b,h) (16 blocks, 128 threads).
// Thread t owns C row v=t (128 floats in VGPRs) and n[t]. Per-step q/k/v is
// double-buffered into LDS with gfx1250 async global->LDS loads so the next
// step's fetch overlaps the current 128-deep FMA chain; DH^-0.5 is folded
// into i_act so raw k can be DMA'd. Falls back to sync staging if the async
// builtins are unavailable.
// ---------------------------------------------------------------------------
__global__ __launch_bounds__(128) void scan_kernel(
    const float* __restrict__ qkv, const float* __restrict__ ipre,
    const float* __restrict__ logfg, float* __restrict__ hbuf,
    float* __restrict__ Cout, float* __restrict__ nout,
    float* __restrict__ mout) {
  const int b = blockIdx.x >> 3;
  const int h = blockIdx.x & 7;
  const int t = threadIdx.x;           // 0..127
  const int lane = t & 31, wid = t >> 5;
  __shared__ float buf[2][384];        // [stage][ q:0..127 | k:128..255 | v:256..383 ]
  __shared__ float red[4];
  float Crow[128];
#pragma unroll
  for (int c = 0; c < 128; ++c) Crow[c] = 0.f;
  float n_t = 0.f, m = 0.f;
  const float kscale = 0.08838834764831845f;   // DH^-0.5

#if USE_ASYNC_LDS
  {
    const size_t b0 = (size_t)(b * 512) * 3072 + (size_t)h * 128 + t;
    __builtin_amdgcn_global_load_async_to_lds_b32(
        (GAS int*)(qkv + b0), (LAS int*)&buf[0][t], 0, 0);
    __builtin_amdgcn_global_load_async_to_lds_b32(
        (GAS int*)(qkv + b0 + 1024), (LAS int*)&buf[0][128 + t], 0, 0);
    __builtin_amdgcn_global_load_async_to_lds_b32(
        (GAS int*)(qkv + b0 + 2048), (LAS int*)&buf[0][256 + t], 0, 0);
  }
#endif

  for (int s = 0; s < 512; ++s) {
    const float* sb;
#if USE_ASYNC_LDS
    __builtin_amdgcn_s_wait_asynccnt(0);   // this wave's stage-(s) DMA done
    __syncthreads();                        // all waves' stage-(s) data visible
    if (s + 1 < 512) {                      // kick off stage-(s+1) DMA
      const size_t bn = (size_t)(b * 512 + s + 1) * 3072 + (size_t)h * 128 + t;
      float* nb = buf[(s + 1) & 1];
      __builtin_amdgcn_global_load_async_to_lds_b32(
          (GAS int*)(qkv + bn), (LAS int*)&nb[t], 0, 0);
      __builtin_amdgcn_global_load_async_to_lds_b32(
          (GAS int*)(qkv + bn + 1024), (LAS int*)&nb[128 + t], 0, 0);
      __builtin_amdgcn_global_load_async_to_lds_b32(
          (GAS int*)(qkv + bn + 2048), (LAS int*)&nb[256 + t], 0, 0);
    }
    sb = buf[s & 1];
#else
    const size_t base = (size_t)(b * 512 + s) * 3072 + (size_t)h * 128;
    buf[0][t]       = qkv[base + t];
    buf[0][128 + t] = qkv[base + 1024 + t];
    buf[0][256 + t] = qkv[base + 2048 + t];
    __syncthreads();
    sb = buf[0];
#endif
    const int gi = (b * 512 + s) * 8 + h;
    const float it    = ipre[gi];
    const float lft   = logfg[gi];
    const float m_new = fmaxf(lft + m, it);
    const float f_act = expf(lft + m - m_new);
    const float iks   = expf(it - m_new) * kscale;  // i_act with k-scale folded
    m = m_new;
    n_t = fmaf(f_act, n_t, iks * sb[128 + t]);
    const float ivr = iks * sb[256 + t];
    float hnum = 0.f;
#pragma unroll
    for (int c = 0; c < 128; ++c) {
      Crow[c] = fmaf(f_act, Crow[c], ivr * sb[128 + c]);
      hnum = fmaf(Crow[c], sb[c], hnum);
    }
    float qn = n_t * sb[t];
#pragma unroll
    for (int off = 16; off > 0; off >>= 1) qn += __shfl_xor(qn, off, 32);
    if (lane == 0) red[wid] = qn;
    __syncthreads();
    qn = red[0] + red[1] + red[2] + red[3];
    hbuf[(size_t)(b * 512 + s) * 1024 + h * 128 + t] =
        hnum / fmaxf(fabsf(qn), 1.f);
#if !USE_ASYNC_LDS
    __syncthreads();   // buf[0] reused next step
#endif
  }
  // Final states -> d_out tail
  float* Cp = Cout + ((size_t)(b * 8 + h) * 128 + t) * 128;
#pragma unroll
  for (int c = 0; c < 128; ++c) Cp[c] = Crow[c];
  nout[(size_t)(b * 8 + h) * 128 + t] = n_t;
  if (t == 0) mout[b * 8 + h] = m;
}

// ---------------------------------------------------------------------------
// Per-head weight-only LayerNorm + learnable skip of xa + silu(z) gating.
// Block per (b,s); wave w handles head w (128 elems, 4 per lane).
// ---------------------------------------------------------------------------
__global__ __launch_bounds__(256) void post_kernel(
    const float* __restrict__ hbuf, const float* __restrict__ hh,
    const float* __restrict__ xa, const float* __restrict__ ln_w,
    const float* __restrict__ skip, float* __restrict__ hstate) {
  const int bs   = blockIdx.x;
  const int head = threadIdx.x >> 5;
  const int lane = threadIdx.x & 31;
  const float* hp = hbuf + (size_t)bs * 1024 + head * 128;
  float v[4];
  float s1 = 0.f, s2 = 0.f;
#pragma unroll
  for (int j = 0; j < 4; ++j) {
    v[j] = hp[lane * 4 + j];
    s1 += v[j];
    s2 = fmaf(v[j], v[j], s2);
  }
#pragma unroll
  for (int off = 16; off > 0; off >>= 1) {
    s1 += __shfl_xor(s1, off, 32);
    s2 += __shfl_xor(s2, off, 32);
  }
  const float mu   = s1 * (1.f / 128.f);
  const float var  = s2 * (1.f / 128.f) - mu * mu;
  const float rstd = rsqrtf(var + 1e-6f);
#pragma unroll
  for (int j = 0; j < 4; ++j) {
    const int c    = head * 128 + lane * 4 + j;
    const float hl = (v[j] - mu) * rstd * ln_w[c];
    const float hs = hl + skip[c] * xa[(size_t)bs * 1024 + c];
    const float z  = hh[(size_t)bs * 2048 + 1024 + c];
    hstate[(size_t)bs * 1024 + c] = hs * (z / (1.f + expf(-z)));
  }
}

// ---------------------------------------------------------------------------
extern "C" void kernel_launch(void* const* d_in, const int* in_sizes, int n_in,
                              void* d_out, int out_size, void* d_ws, size_t ws_size,
                              hipStream_t stream) {
  (void)in_sizes; (void)n_in; (void)out_size; (void)ws_size;
  const float* x      = (const float*)d_in[0];
  const float* w_up   = (const float*)d_in[1];
  const float* conv_w = (const float*)d_in[2];
  const float* conv_b = (const float*)d_in[3];
  const float* w_q    = (const float*)d_in[4];
  const float* w_k    = (const float*)d_in[5];
  const float* w_v    = (const float*)d_in[6];
  const float* w_i    = (const float*)d_in[7];
  const float* b_i    = (const float*)d_in[8];
  const float* w_f    = (const float*)d_in[9];
  const float* b_f    = (const float*)d_in[10];
  const float* ln_w   = (const float*)d_in[11];
  const float* skip   = (const float*)d_in[12];
  const float* w_down = (const float*)d_in[13];

  float* ws = (float*)d_ws;
  float* hh     = ws;               // (1024, 2048)               2,097,152
  float* qkv    = hh     + 2097152; // (1024, 3072) [q|k|v]       3,145,728
  float* xa     = qkv    + 3145728; // (1024, 1024)               1,048,576
  float* ipre   = xa     + 1048576; // (1024, 8)                      8,192
  float* logfg  = ipre   + 8192;    // (1024, 8)                      8,192
  float* hbuf   = logfg  + 8192;    // (1024, 1024)               1,048,576
  float* hstate = hbuf   + 1048576; // (1024, 1024)               1,048,576

  float* y    = (float*)d_out;      // (2,512,512)  = 524288
  float* Cout = y    + 524288;      // (2,8,128,128)= 262144
  float* nout = Cout + 262144;      // (2,8,128)    =   2048
  float* mout = nout + 2048;        // (2,8)        =     16

  // 1) hh = x @ w_up (M=1024,N=2048,K=512): 64*32=2048 strips / 8 waves/block
  wmma_gemm_kernel<<<256, 256, 0, stream>>>(x, w_up, hh, 1024, 2048, 512);
  // 2) causal conv + SiLU + block-diag q/k/v
  conv_qkv_kernel<<<1024, 256, 0, stream>>>(hh, conv_w, conv_b, w_q, w_k, w_v,
                                            qkv, xa);
  // 3) gate pre-activations
  gates_kernel<<<1024, 512, 0, stream>>>(qkv, w_i, b_i, w_f, b_f, ipre, logfg);
  // 4) sequential mLSTM scan (also writes final C, n, m)
  scan_kernel<<<16, 128, 0, stream>>>(qkv, ipre, logfg, hbuf, Cout, nout, mout);
  // 5) LN + skip + silu(z) gating
  post_kernel<<<1024, 256, 0, stream>>>(hbuf, hh, xa, ln_w, skip, hstate);
  // 6) y = hstate @ w_down (M=1024,N=512,K=1024): 64*8=512 strips / 64 blocks
  wmma_gemm_kernel<<<64, 256, 0, stream>>>(hstate, w_down, y, 1024, 512, 1024);
}
